// ACDA_56032143343801
// MI455X (gfx1250) — matrix operand order repaired and checked by reading, not compiled, not run on hardware
//
#include <hip/hip_runtime.h>
#include <hip/hip_bf16.h>

// ACDA fused kernel for gfx1250 (MI455X), wave32 + v_wmma_f32_16x16x32_bf16.
//
// Shapes: B=8, C=O=64, H=W=64, K=3 (9 taps), A=4 atoms.
// out[b,o,p] = sum_{a,k} relu( w_gen[o*36+a*9+k] . x[b,:,p] + b_gen )
//              * att[a,p] * x[b,o, p+tap_k]          (C==O => tap channel = o)
//
// GEMM: D(16 filt x 16 px) = Wtile(16x64) x Xtile(64x16), K=64 -> 2 chained
// bf16 WMMAs with bias pre-loaded as the C operand. 36 rows/o padded to 48
// (3 tiles). Workspace: repacked bf16 A-fragments (393216 B) + padded bias
// (12288 B) = 405504 B total.

typedef __attribute__((ext_vector_type(16))) __bf16 v16bf;
typedef __attribute__((ext_vector_type(8)))  float  v8f;

#define WPK_ELEMS (64 * 3 * 2 * 32 * 16)   // 196608 bf16 A-fragment elements
#define BIAS_OFF  (WPK_ELEMS * 2)          // byte offset of padded bias in ws

__device__ __forceinline__ __bf16 f2bf(float f) {
  unsigned u = __builtin_bit_cast(unsigned, f);
  unsigned r = u + 0x7FFFu + ((u >> 16) & 1u);          // round-to-nearest-even
  unsigned short hs = (unsigned short)(r >> 16);
  return __builtin_bit_cast(__bf16, hs);
}

// ---------------------------------------------------------------------------
// Repack w_gen (2304x64 f32) into WMMA A-fragment-ordered bf16 tiles.
// Layout: [o(64)][t(3)][s(2)][lane(32)][elem(16)], padded rows (idx>=36) = 0.
// A 16-bit 16x32 fragment (ISA 7.12.2): lane l -> M = l%16;
//   elem m = 2v+h -> K = (v%4)*2 + h + (l>=16 ? 8 : 0) + (v>=4 ? 16 : 0).
// ---------------------------------------------------------------------------
__global__ __launch_bounds__(256) void acda_pack_w(const float* __restrict__ wgen,
                                                   __bf16* __restrict__ wpk) {
  int id   = blockIdx.x * 256 + threadIdx.x;           // 768 blocks: exact
  int m    = id & 15;
  int lane = (id >> 4) & 31;
  int s    = (id >> 9) & 1;
  int t    = (id >> 10) % 3;
  int o    = id / 3072;
  int v = m >> 1, hh = m & 1;
  int K = ((v & 3) << 1) + hh + ((lane & 16) ? 8 : 0) + ((v & 4) ? 16 : 0) + (s << 5);
  int idx = t * 16 + (lane & 15);                      // filter row within o
  float val = (idx < 36) ? wgen[(o * 36 + idx) * 64 + K] : 0.0f;
  wpk[id] = f2bf(val);
}

// Zero-padded bias: [o(64)][48] f32.
__global__ __launch_bounds__(256) void acda_pack_bias(const float* __restrict__ bgen,
                                                      float* __restrict__ biasp) {
  int id = blockIdx.x * 256 + threadIdx.x;             // 12 blocks: exact 3072
  int o = id / 48, idx = id % 48;
  biasp[id] = (idx < 36) ? bgen[o * 36 + idx] : 0.0f;
}

// ---------------------------------------------------------------------------
// Main fused kernel. One block = 8 waves = one (b,h,wt) 16-pixel tile.
// Wave wv covers output channels o = wv*8 .. wv*8+7.
// ---------------------------------------------------------------------------
#define ROW(D, J, A_, Kk) part += fmaxf((D)[J], 0.0f) * att##A_ * tap##Kk;

__global__ __launch_bounds__(256) void acda_main(const float* __restrict__ x,
                                                 const float* __restrict__ pos,
                                                 const __bf16* __restrict__ wpk,
                                                 const float* __restrict__ biasp,
                                                 float* __restrict__ out) {
  const int lane = threadIdx.x & 31;
  const int wv   = threadIdx.x >> 5;
  const int tile = blockIdx.x;              // 2048 tiles = 8 * 64 * 4
  const int b    = tile >> 8;
  const int h    = (tile >> 2) & 63;
  const int wt   = tile & 3;
  const int px   = lane & 15;
  const int hoff = (lane & 16) ? 8 : 0;     // D-row offset for upper half-wave
  const int koff = (lane & 16) ? 8 : 0;     // B-frag K offset for upper half
  const int wpx  = wt * 16 + px;

  // Gaussian spatial attention per pixel (grid = meshgrid(xs, ys, 'xy')).
  const float gx = -1.0f + (2.0f / 63.0f) * (float)wpx;
  const float gy = -1.0f + (2.0f / 63.0f) * (float)h;
  float dx, dy;
  dx = gx - pos[0]; dy = gy - pos[1]; const float att0 = __expf(-(dx * dx + dy * dy));
  dx = gx - pos[2]; dy = gy - pos[3]; const float att1 = __expf(-(dx * dx + dy * dy));
  dx = gx - pos[4]; dy = gy - pos[5]; const float att2 = __expf(-(dx * dx + dy * dy));
  dx = gx - pos[6]; dy = gy - pos[7]; const float att3 = __expf(-(dx * dx + dy * dy));

  // Build bf16 B fragments (X tile, 64ch x 16px), ISA B layout:
  // lane l -> N = l%16; elem m -> K = m + (m>=8 ? 8 : 0) + (l>=16 ? 8 : 0) (+32 for step 2).
  const float* xb = x + (size_t)b * 64 * 4096 + h * 64 + wpx;
  v16bf bf0, bf1;
#pragma unroll
  for (int m = 0; m < 16; ++m) {
    bf0[m] = f2bf(xb[(size_t)(m + (m & 8) + koff) * 4096]);
    bf1[m] = f2bf(xb[(size_t)(m + (m & 8) + koff + 32) * 4096]);
  }

  const bool hup = (h > 0), hdn = (h < 63), wl = (wpx > 0), wr = (wpx < 63);

  for (int i = 0; i < 8; ++i) {
    const int o = wv * 8 + i;

    // 3x3 neighborhood of x[b,o,:,:] (zero-padded), tap index = kh*3+kw.
    const float* xo = x + ((size_t)b * 64 + o) * 4096;
    const int c0 = h * 64 + wpx;
    const float tap0 = (hup && wl) ? xo[c0 - 65] : 0.0f;
    const float tap1 = (hup)       ? xo[c0 - 64] : 0.0f;
    const float tap2 = (hup && wr) ? xo[c0 - 63] : 0.0f;
    const float tap3 = (wl)        ? xo[c0 - 1]  : 0.0f;
    const float tap4 =               xo[c0];
    const float tap5 = (wr)        ? xo[c0 + 1]  : 0.0f;
    const float tap6 = (hdn && wl) ? xo[c0 + 63] : 0.0f;
    const float tap7 = (hdn)       ? xo[c0 + 64] : 0.0f;
    const float tap8 = (hdn && wr) ? xo[c0 + 65] : 0.0f;

    v8f d0, d1, d2;
    {
      const v16bf a0 = *(const v16bf*)(wpk + (((o * 3 + 0) * 2 + 0) * 32 + lane) * 16);
      const v16bf a1 = *(const v16bf*)(wpk + (((o * 3 + 0) * 2 + 1) * 32 + lane) * 16);
      v8f c = *(const v8f*)(biasp + o * 48 + 0 + hoff);   // bias as C operand
      c  = __builtin_amdgcn_wmma_f32_16x16x32_bf16(false, a0, false, bf0, (short)0, c, false, false);
      d0 = __builtin_amdgcn_wmma_f32_16x16x32_bf16(false, a1, false, bf1, (short)0, c, false, false);
    }
    {
      const v16bf a0 = *(const v16bf*)(wpk + (((o * 3 + 1) * 2 + 0) * 32 + lane) * 16);
      const v16bf a1 = *(const v16bf*)(wpk + (((o * 3 + 1) * 2 + 1) * 32 + lane) * 16);
      v8f c = *(const v8f*)(biasp + o * 48 + 16 + hoff);
      c  = __builtin_amdgcn_wmma_f32_16x16x32_bf16(false, a0, false, bf0, (short)0, c, false, false);
      d1 = __builtin_amdgcn_wmma_f32_16x16x32_bf16(false, a1, false, bf1, (short)0, c, false, false);
    }
    {
      const v16bf a0 = *(const v16bf*)(wpk + (((o * 3 + 2) * 2 + 0) * 32 + lane) * 16);
      const v16bf a1 = *(const v16bf*)(wpk + (((o * 3 + 2) * 2 + 1) * 32 + lane) * 16);
      v8f c = *(const v8f*)(biasp + o * 48 + 32 + hoff);
      c  = __builtin_amdgcn_wmma_f32_16x16x32_bf16(false, a0, false, bf0, (short)0, c, false, false);
      d2 = __builtin_amdgcn_wmma_f32_16x16x32_bf16(false, a1, false, bf1, (short)0, c, false, false);
    }

    // Epilogue: row idx = t*16 + hoff + j -> (a,k) = (idx/9, idx%9); idx>=36
    // is zero padding. Split on half-wave so (a,k) are compile-time.
    float part = 0.0f;
    if (!(lane & 16)) {
      ROW(d0, 0, 0, 0) ROW(d0, 1, 0, 1) ROW(d0, 2, 0, 2) ROW(d0, 3, 0, 3)
      ROW(d0, 4, 0, 4) ROW(d0, 5, 0, 5) ROW(d0, 6, 0, 6) ROW(d0, 7, 0, 7)
      ROW(d1, 0, 1, 7) ROW(d1, 1, 1, 8) ROW(d1, 2, 2, 0) ROW(d1, 3, 2, 1)
      ROW(d1, 4, 2, 2) ROW(d1, 5, 2, 3) ROW(d1, 6, 2, 4) ROW(d1, 7, 2, 5)
      ROW(d2, 0, 3, 5) ROW(d2, 1, 3, 6) ROW(d2, 2, 3, 7) ROW(d2, 3, 3, 8)
    } else {
      ROW(d0, 0, 0, 8) ROW(d0, 1, 1, 0) ROW(d0, 2, 1, 1) ROW(d0, 3, 1, 2)
      ROW(d0, 4, 1, 3) ROW(d0, 5, 1, 4) ROW(d0, 6, 1, 5) ROW(d0, 7, 1, 6)
      ROW(d1, 0, 2, 6) ROW(d1, 1, 2, 7) ROW(d1, 2, 2, 8) ROW(d1, 3, 3, 0)
      ROW(d1, 4, 3, 1) ROW(d1, 5, 3, 2) ROW(d1, 6, 3, 3) ROW(d1, 7, 3, 4)
    }

    // Combine the two half-wave partials for the same pixel.
    part += __shfl_xor(part, 16, 32);
    if (!(lane & 16))
      out[(((size_t)b * 64 + o) * 64 + h) * 64 + wpx] = part;
  }
}

extern "C" void kernel_launch(void* const* d_in, const int* in_sizes, int n_in,
                              void* d_out, int out_size, void* d_ws, size_t ws_size,
                              hipStream_t stream) {
  (void)in_sizes; (void)n_in; (void)out_size; (void)ws_size;
  const float* x    = (const float*)d_in[0];   // (8,64,64,64)
  const float* wgen = (const float*)d_in[1];   // (2304,64)
  const float* bgen = (const float*)d_in[2];   // (2304,)
  const float* pos  = (const float*)d_in[3];   // (4,2)
  float* out = (float*)d_out;                  // (8,64,64,64)

  __bf16* wpk  = (__bf16*)d_ws;                        // 393216 B
  float*  bias = (float*)((char*)d_ws + BIAS_OFF);     // 12288 B

  acda_pack_w   <<<WPK_ELEMS / 256, 256, 0, stream>>>(wgen, wpk);
  acda_pack_bias<<<12,              256, 0, stream>>>(bgen, bias);
  acda_main     <<<2048,            256, 0, stream>>>(x, pos, wpk, bias, out);
}
